// CustomMultiheadAttention_18786186953508
// MI455X (gfx1250) — matrix operand order repaired and checked by reading, not compiled, not run on hardware
//
#include <hip/hip_runtime.h>
#include <cstdint>
#include <cstddef>

typedef __bf16 bf16_t;
typedef uint32_t u32;
typedef __attribute__((ext_vector_type(16))) __bf16 v16bf;
typedef __attribute__((ext_vector_type(8)))  float  v8f;
typedef __attribute__((ext_vector_type(4)))  uint32_t v4u;
typedef __attribute__((ext_vector_type(8)))  int      v8i;
typedef __attribute__((ext_vector_type(4)))  int      v4i;

union Frag16 {
    v16bf v;
    uint4 q[2];
};

// Load a 16-element bf16 WMMA fragment as two 16-byte LDS reads:
// first uint4 at p, second at p + elemOff2 (in bf16 elements).
__device__ __forceinline__ v16bf load_frag2(const bf16_t* p, int elemOff2) {
    Frag16 f;
    f.q[0] = *(const uint4*)(p);
    f.q[1] = *(const uint4*)(p + elemOff2);
    return f.v;
}

__device__ __forceinline__ v8f wmma_bf16(v16bf a, v16bf b, v8f c) {
    // D = A(16x32) * B(32x16) + C, f32 accum
    return __builtin_amdgcn_wmma_f32_16x16x32_bf16(
        /*neg_a=*/false, a, /*neg_b=*/false, b,
        /*c_mod=*/(short)0, c, /*reuse_a=*/false, /*reuse_b=*/false);
}

// --- CDNA5 async global->LDS copy (ASYNCcnt-tracked, bypasses VGPRs) -------
__device__ __forceinline__ uint32_t lds_off32(const void* p) {
    // LDS aperture: low 32 bits of a generic pointer to __shared__ data are
    // the wave-relative LDS byte offset (what VDST of async-to-LDS expects).
    return (uint32_t)(uintptr_t)p;
}

__device__ __forceinline__ void async_b128(uint32_t ldsOff, const void* gaddr) {
    asm volatile("global_load_async_to_lds_b128 %0, %1, off"
                 : : "v"(ldsOff), "v"(gaddr) : "memory");
}

__device__ __forceinline__ void wait_async0() {
    asm volatile("s_wait_asynccnt 0x0" : : : "memory");
}

// --- CDNA5 Tensor Data Mover: 2D tile load, global -> LDS (TENSORcnt) ------
// D# per ISA cdna5 ch.8: group0 = {count/flags, lds_addr, global_addr, type},
// group1 = {mask/data_size, tensor dims, tile dims, dim0 stride}.
// Tile rows are packed consecutively into LDS (row-major [tileY][tileX]).
__device__ __forceinline__ void tdm_load_2d(
    uint32_t ldsOff, const void* gaddr,
    u32 tileX, u32 tileY,          // tile size in elements (X contiguous)
    u32 tensorX, u32 tensorY,      // tensor extent from tile start (OOB check)
    u32 strideX)                   // elements between consecutive rows
{
    const uint64_t ga = (uint64_t)(uintptr_t)gaddr;
    v4u g0;
    g0[0] = 1u;                                            // count=1 (valid), user
    g0[1] = ldsOff;                                        // lds_addr
    g0[2] = (u32)(ga & 0xFFFFFFFFu);                       // global_addr[31:0]
    g0[3] = (u32)((ga >> 32) & 0x1FFFFFFu) | (2u << 30);   // addr[56:32], type=2
    v8i g1;
    g1[0] = (int)(1u << 16);                               // data_size=1 (2 bytes)
    g1[1] = (int)((tensorX & 0xFFFFu) << 16);              // tensor_dim0[15:0]
    g1[2] = (int)((tensorX >> 16) & 0xFFFFu) |
            (int)((tensorY & 0xFFFFu) << 16);              // dim0[31:16], dim1[15:0]
    g1[3] = (int)((tensorY >> 16) & 0xFFFFu) |
            (int)((tileX & 0xFFFFu) << 16);                // dim1[31:16], tile_dim0
    g1[4] = (int)(tileY & 0xFFFFu);                        // tile_dim1 (tile_dim2=0)
    g1[5] = (int)strideX;                                  // dim0_stride[31:0]
    g1[6] = 0;                                             // dim0_stride[47:32], dim1_stride lo
    g1[7] = 0;
    v4i z4 = {0, 0, 0, 0};
#if __clang_major__ >= 23
    v8i z8 = {0, 0, 0, 0, 0, 0, 0, 0};
    __builtin_amdgcn_tensor_load_to_lds(g0, g1, z4, z4, z8, 0);
#else
    __builtin_amdgcn_tensor_load_to_lds(g0, g1, z4, z4, 0);
#endif
}

__device__ __forceinline__ void wait_tensor0() {
    __builtin_amdgcn_s_wait_tensorcnt((short)0);
}

// ---------------------------------------------------------------------------
// Kernel: fp32 -> bf16 bulk convert
// ---------------------------------------------------------------------------
__global__ void cvt_bf16(const float* __restrict__ src, bf16_t* __restrict__ dst, int n) {
    int i = blockIdx.x * blockDim.x + threadIdx.x;
    int stride = gridDim.x * blockDim.x;
    for (; i < n; i += stride) dst[i] = (bf16_t)src[i];
}

// ---------------------------------------------------------------------------
// Kernel: W[k][n] fp32 -> WT[n][k] bf16 (LDS tile transpose)
// ---------------------------------------------------------------------------
__global__ void transpose_cvt(const float* __restrict__ W, bf16_t* __restrict__ WT, int E) {
    __shared__ float t[32][33];
    const int bx = blockIdx.x * 32, by = blockIdx.y * 32;
    const int x = threadIdx.x, y = threadIdx.y;
    for (int i = 0; i < 32; i += 8)
        t[y + i][x] = W[(size_t)(by + y + i) * E + bx + x];
    __syncthreads();
    for (int i = 0; i < 32; i += 8)
        WT[(size_t)(bx + y + i) * E + by + x] = (bf16_t)t[x][y + i];
}

// ---------------------------------------------------------------------------
// WMMA GEMM: C[M,Nc] = A[M,K](bf16) * BT[Nc,K]^T(bf16) + bias
//   mode 0: fp32 out (final projection)
//   mode 1: per-head L2 row-normalize, bf16 out [B,H,Nseq,64]  (Q / K)
//   mode 2: bf16 out transposed [B,H,64,Nseq]                  (V)
// Block: 128x128 tile, 256 threads = 8 waves (4 m-waves x 2 n-waves).
// Double-buffered TDM (tensor_load_to_lds) staging, issued by wave 0.
// ---------------------------------------------------------------------------
__global__ __launch_bounds__(256) void gemm_qkv(
    const bf16_t* __restrict__ A,
    const bf16_t* __restrict__ BT,
    const float*  __restrict__ bias,
    float*        __restrict__ outF,
    bf16_t*       __restrict__ outB,
    int M, int Nc, int K, int mode)
{
    __shared__ __align__(16) bf16_t Alds[2][128 * 32];
    __shared__ __align__(16) bf16_t Blds[2][128 * 32];

    const int tid  = threadIdx.x;
    const int wave = tid >> 5;
    const int lane = tid & 31;
    const int g    = lane >> 4;   // half-wave group
    const int ln   = lane & 15;
    const int wm   = wave >> 1;   // 0..3 -> 32-row strip
    const int wn   = wave & 1;    // 0..1 -> 64-col strip
    const int blockM = blockIdx.y * 128;
    const int blockN = blockIdx.x * 128;

    auto prefetch = [&](int buf, int k0) {
        if (wave == 0) {   // TDM ignores EXEC; one DMA per tile via scalar branch
            tdm_load_2d(lds_off32(&Alds[buf][0]),
                        A + (size_t)blockM * K + k0,
                        /*tileX=*/32, /*tileY=*/128,
                        /*tensorX=*/(u32)(K - k0), /*tensorY=*/128, /*strideX=*/(u32)K);
            tdm_load_2d(lds_off32(&Blds[buf][0]),
                        BT + (size_t)blockN * K + k0,
                        32, 128, (u32)(K - k0), 128, (u32)K);
        }
    };

    v8f acc[2][4] = {};

    prefetch(0, 0);
    for (int k0 = 0; k0 < K; k0 += 32) {
        const int cur = (k0 >> 5) & 1;
        if (wave == 0) wait_tensor0();  // DMA into `cur` complete
        __syncthreads();                // publish LDS; prev reads of cur^1 done
        if (k0 + 32 < K) prefetch(cur ^ 1, k0 + 32);

        v16bf bfr[4];
        for (int nt = 0; nt < 4; ++nt) {
            const int n = wn * 64 + nt * 16 + ln;
            bfr[nt] = load_frag2(&Blds[cur][n * 32 + g * 16], 8);
        }
        for (int mt = 0; mt < 2; ++mt) {
            const int m = wm * 32 + mt * 16 + ln;
            v16bf af = load_frag2(&Alds[cur][m * 32 + g * 16], 8);
            for (int nt = 0; nt < 4; ++nt)
                acc[mt][nt] = wmma_bf16(af, bfr[nt], acc[mt][nt]);
        }
    }

    float bv[4];
    for (int nt = 0; nt < 4; ++nt)
        bv[nt] = bias[blockN + wn * 64 + nt * 16 + ln];

    if (mode == 0) {
        for (int mt = 0; mt < 2; ++mt)
            for (int nt = 0; nt < 4; ++nt) {
                const int n = blockN + wn * 64 + nt * 16 + ln;
                for (int r = 0; r < 8; ++r) {
                    const int m = blockM + wm * 32 + mt * 16 + g * 8 + r;
                    outF[(size_t)m * Nc + n] = acc[mt][nt][r] + bv[nt];
                }
            }
        return;
    }

    // Wave's 64 columns == exactly one head (blockN, wn are 64-aligned).
    const int h = (blockN + wn * 64) >> 6;
    for (int mt = 0; mt < 2; ++mt) {
        for (int nt = 0; nt < 4; ++nt)
            for (int r = 0; r < 8; ++r)
                acc[mt][nt][r] += bv[nt];

        for (int r = 0; r < 8; ++r) {
            const int m  = blockM + wm * 32 + mt * 16 + g * 8 + r;
            const int b  = m >> 11;       // / 2048 (seq len)
            const int ns = m & 2047;
            if (mode == 1) {
                float ss = 0.f;
                for (int nt = 0; nt < 4; ++nt) {
                    const float t = acc[mt][nt][r];
                    ss += t * t;
                }
                ss += __shfl_xor(ss, 1, 32);
                ss += __shfl_xor(ss, 2, 32);
                ss += __shfl_xor(ss, 4, 32);
                ss += __shfl_xor(ss, 8, 32);
                const float inv = rsqrtf(ss + 1e-12f);
                for (int nt = 0; nt < 4; ++nt) {
                    const int d = nt * 16 + ln;
                    outB[(((size_t)b * 16 + h) * 2048 + ns) * 64 + d] =
                        (bf16_t)(acc[mt][nt][r] * inv);
                }
            } else { // mode 2: V, transposed [B,H,64,N]
                for (int nt = 0; nt < 4; ++nt) {
                    const int d = nt * 16 + ln;
                    outB[(((size_t)b * 16 + h) * 64 + d) * 2048 + ns] =
                        (bf16_t)acc[mt][nt][r];
                }
            }
        }
    }
}

// ---------------------------------------------------------------------------
// Flash attention: one block per (b*H + h, 128-query tile).
// 8 waves; wave w owns query rows [16w, 16w+16) end-to-end.
// qn/kn: [BH, 2048, 64] bf16 (L2-normalized). vt: [BH, 64, 2048] bf16.
// Double-buffered async global->LDS K/V staging (ASYNCcnt path).
// 112 KB LDS (< 320 KB WGP budget).
// ---------------------------------------------------------------------------
__global__ __launch_bounds__(256) void attn_fa(
    const bf16_t* __restrict__ qn,
    const bf16_t* __restrict__ kn,
    const bf16_t* __restrict__ vt,
    bf16_t*       __restrict__ outB)
{
    __shared__ __align__(16) bf16_t Qs[128 * 64];        // 16 KB
    __shared__ __align__(16) bf16_t Ks[2][128 * 64];     // 32 KB
    __shared__ __align__(16) bf16_t Vs[2][64 * 128];     // 32 KB [d][key]
    __shared__ __align__(16) bf16_t Ps[128 * 128];       // 32 KB

    const int tid  = threadIdx.x;
    const int wave = tid >> 5;
    const int lane = tid & 31;
    const int g    = lane >> 4;
    const int ln   = lane & 15;

    const int bh = blockIdx.x;          // b*16 + h
    const int qb = blockIdx.y * 128;

    const int row  = tid >> 1, half = tid & 1;   // K/Q tiles: 2 threads/row
    const int vrow = tid >> 2, vq   = tid & 3;   // V tile: 4 threads/row

    const bf16_t* gQ = qn + ((size_t)bh * 2048 + qb + row) * 64 + half * 32;
    const bf16_t* gK = kn + ((size_t)bh * 2048 + row) * 64 + half * 32;
    const bf16_t* gV = vt + ((size_t)bh * 64 + vrow) * 2048 + vq * 32;

    auto prefetchKV = [&](int buf, int kb) {
        const uint32_t lk = lds_off32(&Ks[buf][row * 64 + half * 32]);
        const uint32_t lv = lds_off32(&Vs[buf][vrow * 128 + vq * 32]);
        async_b128(lk,      gK + (size_t)kb * 64);
        async_b128(lk + 16, gK + (size_t)kb * 64 + 8);
        async_b128(lv,      gV + kb);
        async_b128(lv + 16, gV + kb + 8);
    };

    {   // async Q tile load (waited together with first K/V tile)
        const uint32_t lq = lds_off32(&Qs[row * 64 + half * 32]);
        async_b128(lq,      gQ);
        async_b128(lq + 16, gQ + 8);
    }
    prefetchKV(0, 0);

    v8f o[4] = {};
    float mOld[8], lSum[8];
    for (int r = 0; r < 8; ++r) { mOld[r] = -1e30f; lSum[r] = 0.f; }

    const int mrow = wave * 16 + ln;    // this lane's A-fragment row

    for (int it = 0; it < 16; ++it) {
        const int kb  = it * 128;
        const int cur = it & 1;
        wait_async0();
        __syncthreads();
        if (it + 1 < 16) prefetchKV(cur ^ 1, kb + 128);

        // S = Qn * Kn^T  (16 rows x 128 keys per wave)
        v16bf qa0 = load_frag2(&Qs[mrow * 64 +  0 + g * 16], 8);
        v16bf qa1 = load_frag2(&Qs[mrow * 64 + 32 + g * 16], 8);
        v8f s[8];
        for (int nt = 0; nt < 8; ++nt) {
            const int n = nt * 16 + ln;
            v16bf b0 = load_frag2(&Ks[cur][n * 64 +  0 + g * 16], 8);
            v16bf b1 = load_frag2(&Ks[cur][n * 64 + 32 + g * 16], 8);
            v8f c = {};
            c = wmma_bf16(qa0, b0, c);
            c = wmma_bf16(qa1, b1, c);
            s[nt] = c;
        }

        // Online softmax (stats per owned row r; lanes within a 16-group share)
        float scale[8];
        for (int r = 0; r < 8; ++r) {
            float mx = s[0][r];
            for (int nt = 1; nt < 8; ++nt) mx = fmaxf(mx, s[nt][r]);
            mx = fmaxf(mx, __shfl_xor(mx, 1, 32));
            mx = fmaxf(mx, __shfl_xor(mx, 2, 32));
            mx = fmaxf(mx, __shfl_xor(mx, 4, 32));
            mx = fmaxf(mx, __shfl_xor(mx, 8, 32));
            const float mNew = fmaxf(mOld[r], mx);
            scale[r] = __expf(mOld[r] - mNew);
            mOld[r]  = mNew;
        }
        for (int nt = 0; nt < 8; ++nt)
            for (int r = 0; r < 8; ++r)
                s[nt][r] = __expf(s[nt][r] - mOld[r]);
        for (int r = 0; r < 8; ++r) {
            float rs = 0.f;
            for (int nt = 0; nt < 8; ++nt) rs += s[nt][r];
            rs += __shfl_xor(rs, 1, 32);
            rs += __shfl_xor(rs, 2, 32);
            rs += __shfl_xor(rs, 4, 32);
            rs += __shfl_xor(rs, 8, 32);
            lSum[r] = lSum[r] * scale[r] + rs;
        }
        for (int dt = 0; dt < 4; ++dt)
            for (int r = 0; r < 8; ++r)
                o[dt][r] *= scale[r];

        // P -> LDS (re-layout C-frag to A-frag; wave-private region, DS in-order)
        for (int nt = 0; nt < 8; ++nt)
            for (int r = 0; r < 8; ++r)
                Ps[(wave * 16 + g * 8 + r) * 128 + nt * 16 + ln] = (bf16_t)s[nt][r];

        // O += P * V
        for (int kk = 0; kk < 4; ++kk) {
            v16bf pa = load_frag2(&Ps[mrow * 128 + kk * 32 + g * 16], 8);
            for (int dt = 0; dt < 4; ++dt) {
                const int d = dt * 16 + ln;
                v16bf vb = load_frag2(&Vs[cur][d * 128 + kk * 32 + g * 16], 8);
                o[dt] = wmma_bf16(pa, vb, o[dt]);
            }
        }
    }

    // Normalize and write merged-head bf16 output [B, 2048, 1024]
    const int b = bh >> 4, h = bh & 15;
    for (int r = 0; r < 8; ++r) {
        const float inv = 1.f / lSum[r];
        const int qm = qb + wave * 16 + g * 8 + r;
        for (int dt = 0; dt < 4; ++dt)
            outB[((size_t)b * 2048 + qm) * 1024 + h * 64 + dt * 16 + ln] =
                (bf16_t)(o[dt][r] * inv);
    }
}

// ---------------------------------------------------------------------------
extern "C" void kernel_launch(void* const* d_in, const int* in_sizes, int n_in,
                              void* d_out, int out_size, void* d_ws, size_t ws_size,
                              hipStream_t stream)
{
    (void)in_sizes; (void)n_in; (void)out_size; (void)ws_size;

    const float* x    = (const float*)d_in[0];
    const float* Wq_w = (const float*)d_in[1];
    const float* Wq_b = (const float*)d_in[2];
    const float* Wk_w = (const float*)d_in[3];
    const float* Wk_b = (const float*)d_in[4];
    const float* Wv_w = (const float*)d_in[5];
    const float* Wv_b = (const float*)d_in[6];
    const float* Wo_w = (const float*)d_in[7];
    const float* Wo_b = (const float*)d_in[8];
    float* out = (float*)d_out;

    const int Bc = 4, Nn = 2048, E = 1024, H = 16;
    const int M = Bc * Nn;  // 8192

    char* ws = (char*)d_ws;
    size_t off = 0;
    bf16_t* xbf = (bf16_t*)(ws + off); off += (size_t)M * E * 2;
    bf16_t* WqT = (bf16_t*)(ws + off); off += (size_t)E * E * 2;
    bf16_t* WkT = (bf16_t*)(ws + off); off += (size_t)E * E * 2;
    bf16_t* WvT = (bf16_t*)(ws + off); off += (size_t)E * E * 2;
    bf16_t* WoT = (bf16_t*)(ws + off); off += (size_t)E * E * 2;
    bf16_t* qn  = (bf16_t*)(ws + off); off += (size_t)M * E * 2;
    bf16_t* kn  = (bf16_t*)(ws + off); off += (size_t)M * E * 2;
    bf16_t* vtb = (bf16_t*)(ws + off); off += (size_t)M * E * 2;
    bf16_t* abf = (bf16_t*)(ws + off); off += (size_t)M * E * 2;

    cvt_bf16<<<4096, 256, 0, stream>>>(x, xbf, M * E);

    dim3 tb(32, 8);
    dim3 tg(E / 32, E / 32);
    transpose_cvt<<<tg, tb, 0, stream>>>(Wq_w, WqT, E);
    transpose_cvt<<<tg, tb, 0, stream>>>(Wk_w, WkT, E);
    transpose_cvt<<<tg, tb, 0, stream>>>(Wv_w, WvT, E);
    transpose_cvt<<<tg, tb, 0, stream>>>(Wo_w, WoT, E);

    dim3 gg(E / 128, M / 128);  // (8, 64)
    gemm_qkv<<<gg, 256, 0, stream>>>(xbf, WqT, Wq_b, nullptr, qn,  M, E, E, 1);
    gemm_qkv<<<gg, 256, 0, stream>>>(xbf, WkT, Wk_b, nullptr, kn,  M, E, E, 1);
    gemm_qkv<<<gg, 256, 0, stream>>>(xbf, WvT, Wv_b, nullptr, vtb, M, E, E, 2);

    dim3 ag(Bc * H, Nn / 128);  // (64, 16)
    attn_fa<<<ag, 256, 0, stream>>>(qn, kn, vtb, abf);

    gemm_qkv<<<gg, 256, 0, stream>>>(abf, WoT, Wo_b, out, nullptr, M, E, E, 0);
}
